// FBSNN_89996744720432
// MI455X (gfx1250) — matrix operand order repaired and checked by reading, not compiled, not run on hardware
//
#include <hip/hip_runtime.h>
#include <hip/hip_bf16.h>
#include <math.h>

typedef __attribute__((ext_vector_type(16))) _Float16 v16h;
typedef __attribute__((ext_vector_type(8)))  _Float16 v8h;
typedef __attribute__((ext_vector_type(8)))  float    v8f;

#define Mtot   1024
#define NSTEP  50
#define Dd     512
#define Hh     256
#define MT     16
#define TPB    256

// packed weight regions (offsets in halves). Layout per region: blocks of
// (kb,nb) 32x16; within a block lane l (N = l%16 within tile) holds 16
// halves (K within tile, order 8h+s / 16+8h+s-8) -> exactly one WMMA B-fragment per 32B lane load.
#define OFF_B1F 0         // K=512,N=256  B[k][n]=W1[n][k+1]   (X @ W1x^T)
#define OFF_B2F 131072    // K=256,N=256  B[k][n]=W2[n][k]
#define OFF_B3F 196608    //              W3^T
#define OFF_B4F 262144    //              W4^T
#define OFF_B4B 327680    //              B[k][n]=W4[k][n]     (v4 @ W4)
#define OFF_B3B 393216    //              W3
#define OFF_B2B 458752    //              W2
#define OFF_BDU 524288    // K=256,N=512  B[k][n]=W1[k][n+1]   (v1 @ W1x)
#define PACK_TOTAL 655360

__global__ void zero_out_kernel(float* out) {
    if (threadIdx.x == 0 && blockIdx.x == 0) out[0] = 0.0f;
}

__global__ void pack_weights(const float* __restrict__ W1, const float* __restrict__ W2,
                             const float* __restrict__ W3, const float* __restrict__ W4,
                             _Float16* __restrict__ pack) {
    int id = blockIdx.x * blockDim.x + threadIdx.x;
    if (id >= PACK_TOTAL) return;
    int rel, N, mode; const float* S = nullptr;
    if      (id < OFF_B2F) { rel = id - OFF_B1F; N = 256; mode = 2; }
    else if (id < OFF_B3F) { rel = id - OFF_B2F; N = 256; mode = 0; S = W2; }
    else if (id < OFF_B4F) { rel = id - OFF_B3F; N = 256; mode = 0; S = W3; }
    else if (id < OFF_B4B) { rel = id - OFF_B4F; N = 256; mode = 0; S = W4; }
    else if (id < OFF_B3B) { rel = id - OFF_B4B; N = 256; mode = 1; S = W4; }
    else if (id < OFF_B2B) { rel = id - OFF_B3B; N = 256; mode = 1; S = W3; }
    else if (id < OFF_BDU) { rel = id - OFF_B2B; N = 256; mode = 1; S = W2; }
    else                   { rel = id - OFF_BDU; N = 512; mode = 3; }
    int s16 = rel & 15, ln = (rel >> 4) & 31, blk = rel >> 9;
    int nl = ln & 15, kl = ((ln >> 4) << 3) + (s16 & 7) + ((s16 >> 3) << 4);
    int NT = N >> 4;
    int nb = blk % NT, kb = blk / NT;
    int k = kb * 32 + kl, n = nb * 16 + nl;
    float v;
    if      (mode == 0) v = S[n * 256 + k];        // W^T as B
    else if (mode == 1) v = S[k * 256 + n];        // W as B
    else if (mode == 2) v = W1[n * 513 + k + 1];   // W1x^T as B
    else                v = W1[k * 513 + n + 1];   // W1x as B
    pack[id] = (_Float16)v;
}

// ---- WMMA fragment helpers (wave32, 16x16x32 f16 -> f32) ----

// A fragment (16x32 f16) from LDS row-major activations (row stride Dd halves).
// Per ISA: lanes0-15 row M=lane, e0..7 => K=lh*8+e, e8..15 => K=16+lh*8+(e-8).
__device__ __forceinline__ v16h load_A(const _Float16* sH, int lane, int kbase) {
    int lr = lane & 15, lh = lane >> 4;
    const _Float16* p = sH + lr * Dd + kbase + lh * 8;
    v8h lo = *(const v8h*)(p);
    v8h hi = *(const v8h*)(p + 16);
    v16h a;
#pragma unroll
    for (int i = 0; i < 8; ++i) { a[i] = lo[i]; a[i + 8] = hi[i]; }
    return a;
}

// B fragment from the packed weights: force global address space so the
// weight stream is global_load_b128 (LOADcnt only), not flat_load (which
// also ties up DScnt / the LDS datapath on CDNA5).
typedef __attribute__((address_space(1))) const v16h gcv16h;
__device__ __forceinline__ v16h load_B(const v16h* B, int idx) {
    return ((gcv16h*)B)[idx];
}

__device__ __forceinline__ v8f wmma16(v16h a, v16h b, v8f c) {
    return __builtin_amdgcn_wmma_f32_16x16x32_f16(false, a, false, b, (short)0, c,
                                                  false, false);
}

// 2-N-tile GEMM: C(16 x 2*16) += A(16xK) * B(KxN), K = ktiles*32
__device__ __forceinline__ void gemm2(const _Float16* sA, const v16h* B,
                                      int ktiles, int ntiles, int nb0, int nb1,
                                      int lane, v8f& c0, v8f& c1) {
    v8f z = {};
    c0 = z; c1 = z;
    for (int kb = 0; kb < ktiles; ++kb) {
        v16h a  = load_A(sA, lane, kb * 32);
        v16h b0 = load_B(B, (kb * ntiles + nb0) * 32 + lane);
        v16h b1 = load_B(B, (kb * ntiles + nb1) * 32 + lane);
        c0 = wmma16(a, b0, c0);
        c1 = wmma16(a, b1, c1);
    }
}

// forward epilogue: h = sin(z+bias[+t*W1col0]) into dst buffer, stash cos(z)
__device__ __forceinline__ void epi_sin(_Float16* dst, _Float16* sC, int slot,
                                        v8f c0, v8f c1, int wave, int lane,
                                        const float* bias, const float* W1raw, float t1) {
    int lr = lane & 15, lh = lane >> 4;
#pragma unroll
    for (int r = 0; r < 8; ++r) {
        int row = lh * 8 + r;
        int col0 = wave * 16 + lr, col1 = (wave + 8) * 16 + lr;
        float bb0 = bias[col0], bb1 = bias[col1];
        if (W1raw) { bb0 += t1 * W1raw[col0 * 513]; bb1 += t1 * W1raw[col1 * 513]; }
        float z0 = c0[r] + bb0, z1 = c1[r] + bb1;
        dst[row * Dd + col0] = (_Float16)__sinf(z0);
        dst[row * Dd + col1] = (_Float16)__sinf(z1);
        sC[slot * 4096 + row * 256 + col0] = (_Float16)__cosf(z0);
        sC[slot * 4096 + row * 256 + col1] = (_Float16)__cosf(z1);
    }
}

// backward epilogue: v_{l-1} = (v_l @ W_l) * cos(z_{l-1}) into dst buffer
__device__ __forceinline__ void epi_bwd(_Float16* dst, const _Float16* sC, int slot,
                                        v8f c0, v8f c1, int wave, int lane) {
    int lr = lane & 15, lh = lane >> 4;
#pragma unroll
    for (int r = 0; r < 8; ++r) {
        int row = lh * 8 + r;
        int col0 = wave * 16 + lr, col1 = (wave + 8) * 16 + lr;
        dst[row * Dd + col0] = (_Float16)(c0[r] * (float)sC[slot * 4096 + row * 256 + col0]);
        dst[row * Dd + col1] = (_Float16)(c1[r] * (float)sC[slot * 4096 + row * 256 + col1]);
    }
}

// Evaluate u and Du for the block's 16 samples. On entry sH0 holds X (f16,
// 16x512, row stride 512) and all threads are synced. On exit sY holds u,
// sZ holds Du (f32), all threads synced. Activations ping-pong sH0<->sH1 so
// each GEMM stage needs exactly one barrier (epilogue writes the other buf).
__device__ __forceinline__ void net_eval(float t1,
                         float* sZ, _Float16* sH0, _Float16* sH1, _Float16* sC,
                         float* sRedA, float* sY,
                         const v16h* B1F, const v16h* B2F, const v16h* B3F, const v16h* B4F,
                         const v16h* B4B, const v16h* B3B, const v16h* B2B, const v16h* BDU,
                         const float* W1, const float* b1, const float* b2,
                         const float* b3, const float* b4,
                         const float* Wo, const float* bo) {
    int tid = threadIdx.x, wave = tid >> 5, lane = tid & 31;
    v8f c0, c1;

    // layer 1: X(16x512) @ W1x^T -> h1 in sH1
    gemm2(sH0, B1F, 16, 16, wave, wave + 8, lane, c0, c1);
    epi_sin(sH1, sC, 0, c0, c1, wave, lane, b1, W1, t1);
    __syncthreads();
    // layer 2: h1 -> h2 in sH0
    gemm2(sH1, B2F, 8, 16, wave, wave + 8, lane, c0, c1);
    epi_sin(sH0, sC, 1, c0, c1, wave, lane, b2, nullptr, 0.f);
    __syncthreads();
    // layer 3: h2 -> h3 in sH1
    gemm2(sH0, B3F, 8, 16, wave, wave + 8, lane, c0, c1);
    epi_sin(sH1, sC, 2, c0, c1, wave, lane, b3, nullptr, 0.f);
    __syncthreads();
    // layer 4: h3 -> h4 in sH0
    gemm2(sH1, B4F, 8, 16, wave, wave + 8, lane, c0, c1);
    epi_sin(sH0, sC, 3, c0, c1, wave, lane, b4, nullptr, 0.f);
    __syncthreads();

    // u = h4 @ Wo^T + bo (reads sH0); v4 = Wo*cos(z4) staged into sH1
    {
        int row = tid >> 4, c16 = tid & 15;
        float p = 0.f;
#pragma unroll
        for (int j = 0; j < 16; ++j)
            p += (float)sH0[row * Dd + c16 * 16 + j] * Wo[c16 * 16 + j];
        sRedA[row * 16 + c16] = p;
    }
    for (int e = tid; e < MT * Hh; e += TPB) {
        int row = e >> 8, col = e & 255;
        sH1[row * Dd + col] = (_Float16)(Wo[col] * (float)sC[3 * 4096 + row * 256 + col]);
    }
    __syncthreads();
    if (tid < 16) {
        float s = bo[0];
#pragma unroll
        for (int j = 0; j < 16; ++j) s += sRedA[tid * 16 + j];
        sY[tid] = s;
    }

    // backward sweep: v3 in sH0, v2 in sH1, v1 in sH0
    gemm2(sH1, B4B, 8, 16, wave, wave + 8, lane, c0, c1);
    epi_bwd(sH0, sC, 2, c0, c1, wave, lane);
    __syncthreads();
    gemm2(sH0, B3B, 8, 16, wave, wave + 8, lane, c0, c1);
    epi_bwd(sH1, sC, 1, c0, c1, wave, lane);
    __syncthreads();
    gemm2(sH1, B2B, 8, 16, wave, wave + 8, lane, c0, c1);
    epi_bwd(sH0, sC, 0, c0, c1, wave, lane);
    __syncthreads();

    // Du = v1 @ W1x : (16x256) @ (256x512), 4 N-tiles per wave, out to sZ
#pragma unroll
    for (int i = 0; i < 4; ++i) {
        int nb = wave + 8 * i;
        v8f zz = {}; v8f c = zz;
        for (int kb = 0; kb < 8; ++kb) {
            v16h a = load_A(sH0, lane, kb * 32);
            v16h b = load_B(BDU, (kb * 32 + nb) * 32 + lane);
            c = wmma16(a, b, c);
        }
        int lr = lane & 15, lh = lane >> 4;
#pragma unroll
        for (int r = 0; r < 8; ++r) {
            int row = lh * 8 + r;
            sZ[row * Dd + nb * 16 + lr] = c[r];
        }
    }
    __syncthreads();
}

__global__ void __launch_bounds__(TPB)
fbsnn_main(const float* __restrict__ t_g, const float* __restrict__ W_g,
           const float* __restrict__ Xi,
           const float* __restrict__ W1, const float* __restrict__ b1,
           const float* __restrict__ W2, const float* __restrict__ b2,
           const float* __restrict__ W3, const float* __restrict__ b3,
           const float* __restrict__ W4, const float* __restrict__ b4,
           const float* __restrict__ Wo, const float* __restrict__ bo,
           const _Float16* __restrict__ pack, float* out) {
    extern __shared__ char smem[];
    float*    sX    = (float*)smem;                // 16x512 f32  (32 KB)
    float*    sZ    = sX + MT * Dd;                // 16x512 f32  (32 KB)
    _Float16* sH0   = (_Float16*)(sZ + MT * Dd);   // 16x512 f16  (16 KB)
    _Float16* sH1   = sH0 + MT * Dd;               // 16x512 f16  (16 KB)
    _Float16* sC    = sH1 + MT * Dd;               // 4x16x256 f16 (32 KB)
    float*    sRedA = (float*)(sC + 4 * MT * Hh);  // 256 f32
    float*    sRedB = sRedA + 256;                 // 256 f32
    float*    sY    = sRedB + 256;                 // 16 f32
    float*    sYt   = sY + 16;                     // 16 f32

    const v16h* B1F = (const v16h*)(pack + OFF_B1F);
    const v16h* B2F = (const v16h*)(pack + OFF_B2F);
    const v16h* B3F = (const v16h*)(pack + OFF_B3F);
    const v16h* B4F = (const v16h*)(pack + OFF_B4F);
    const v16h* B4B = (const v16h*)(pack + OFF_B4B);
    const v16h* B3B = (const v16h*)(pack + OFF_B3B);
    const v16h* B2B = (const v16h*)(pack + OFF_B2B);
    const v16h* BDU = (const v16h*)(pack + OFF_BDU);

    int tid = threadIdx.x;
    int m0  = blockIdx.x * MT;
    int rowT = tid >> 4, l16 = tid & 15;

    // X0 = Xi broadcast
    for (int e = tid; e < MT * Dd; e += TPB) {
        float x = Xi[e & 511];
        sX[e] = x;
        sH0[e] = (_Float16)x;
    }
    __syncthreads();
    net_eval(t_g[0], sZ, sH0, sH1, sC, sRedA, sY,
             B1F, B2F, B3F, B4F, B4B, B3B, B2B, BDU,
             W1, b1, b2, b3, b4, Wo, bo);

    float lossAcc = 0.f;
    const float* Wrow = W_g + (size_t)(m0 + rowT) * (NSTEP + 1) * Dd;

    for (int n = 1; n <= NSTEP; ++n) {
        float t1  = t_g[n];
        float dt_ = t1 - t_g[n - 1];
        // SDE step + predictor using old (X0,Y0,Z0); update X in place.
        // Brownian increments stream through once -> non-temporal.
        float pXZ = 0.f, pZs = 0.f;
        for (int j = 0; j < 32; ++j) {
            int c = l16 + j * 16;
            int e = rowT * Dd + c;
            float x0 = sX[e], z0 = sZ[e];
            float w1v = __builtin_nontemporal_load(&Wrow[n * Dd + c]);
            float w0v = __builtin_nontemporal_load(&Wrow[(n - 1) * Dd + c]);
            float sdw = 0.4f * x0 * (w1v - w0v);
            pXZ += x0 * z0;
            pZs += z0 * sdw;
            float x1 = x0 + sdw;
            sX[e] = x1;
            sH0[e] = (_Float16)x1;
        }
        sRedA[rowT * 16 + l16] = pXZ;
        sRedB[rowT * 16 + l16] = pZs;
        __syncthreads();
        if (tid < 16) {
            float sXZ = 0.f, sZs = 0.f;
#pragma unroll
            for (int j = 0; j < 16; ++j) { sXZ += sRedA[tid * 16 + j]; sZs += sRedB[tid * 16 + j]; }
            float y0  = sY[tid];
            float phi = 0.05f * (y0 - sXZ);
            sYt[tid]  = y0 + phi * dt_ + sZs;
        }
        __syncthreads();
        net_eval(t1, sZ, sH0, sH1, sC, sRedA, sY,
                 B1F, B2F, B3F, B4F, B4B, B3B, B2B, BDU,
                 W1, b1, b2, b3, b4, Wo, bo);
        if (tid < 16) {
            float d = sY[tid] - sYt[tid];
            lossAcc += d * d;
        }
        __syncthreads();
    }

    // terminal: g = sum(X^2); loss += (Y-g)^2 + sum((Z-2X)^2)
    float pG = 0.f;
    for (int j = 0; j < 32; ++j) {
        int e = rowT * Dd + l16 + j * 16;
        float x = sX[e], z = sZ[e];
        pG += x * x;
        float dz = z - 2.f * x;
        lossAcc += dz * dz;
    }
    sRedA[rowT * 16 + l16] = pG;
    __syncthreads();
    if (tid < 16) {
        float g = 0.f;
#pragma unroll
        for (int j = 0; j < 16; ++j) g += sRedA[tid * 16 + j];
        float d = sY[tid] - g;
        lossAcc += d * d;
    }
    __syncthreads();
    sRedA[tid] = lossAcc;
    __syncthreads();
#pragma unroll
    for (int s = 128; s > 0; s >>= 1) {
        if (tid < s) sRedA[tid] += sRedA[tid + s];
        __syncthreads();
    }
    if (tid == 0) atomicAdd(out, sRedA[0]);
}

extern "C" void kernel_launch(void* const* d_in, const int* in_sizes, int n_in,
                              void* d_out, int out_size, void* d_ws, size_t ws_size,
                              hipStream_t stream) {
    const float* t  = (const float*)d_in[0];
    const float* W  = (const float*)d_in[1];
    const float* Xi = (const float*)d_in[2];
    const float* W1 = (const float*)d_in[3];
    const float* b1 = (const float*)d_in[4];
    const float* W2 = (const float*)d_in[5];
    const float* b2 = (const float*)d_in[6];
    const float* W3 = (const float*)d_in[7];
    const float* b3 = (const float*)d_in[8];
    const float* W4 = (const float*)d_in[9];
    const float* b4 = (const float*)d_in[10];
    const float* Wo = (const float*)d_in[11];
    const float* bo = (const float*)d_in[12];
    _Float16* pack = (_Float16*)d_ws;
    float* out = (float*)d_out;

    zero_out_kernel<<<1, 1, 0, stream>>>(out);
    pack_weights<<<(PACK_TOTAL + 255) / 256, 256, 0, stream>>>(W1, W2, W3, W4, pack);
    size_t smem = (size_t)(MT * Dd) * 4 * 2      // sX, sZ (f32)
                + (size_t)(MT * Dd) * 2 * 2      // sH0, sH1 (f16)
                + (size_t)(4 * MT * Hh) * 2      // sC (f16)
                + (256 + 256 + 16 + 16) * 4;     // reductions
    fbsnn_main<<<Mtot / MT, TPB, smem, stream>>>(t, W, Xi, W1, b1, W2, b2, W3, b3,
                                                 W4, b4, Wo, bo, pack, out);
}
